// Myint_41205916238209
// MI455X (gfx1250) — compile-verified, matrix operand order checked by reading, not run on hardware
//
#include <hip/hip_runtime.h>

typedef __attribute__((ext_vector_type(2)))  float    v2f;
typedef __attribute__((ext_vector_type(8)))  float    v8f;
typedef __attribute__((ext_vector_type(2)))  _Float16 v2h;
typedef __attribute__((ext_vector_type(16))) _Float16 v16h;
typedef __attribute__((ext_vector_type(8)))  unsigned v8u;

static __device__ __forceinline__ float lrelu(float x) { return fmaxf(x, 0.1f * x); }

// pack two f32 -> packed f16 (v_cvt_pk_rtz_f16_f32), re-typed to _Float16 vector
static __device__ __forceinline__ v2h pkrtz(float a, float b) {
  return __builtin_bit_cast(v2h, __builtin_amdgcn_cvt_pkrtz(a, b));
}

static __device__ __forceinline__ v2h lrelu2h(v2h x) {
  v2h s = x * (_Float16)0.1f;                 // v_pk_mul_f16
  return __builtin_elementwise_max(x, s);     // v_pk_max_num_f16
}

static __device__ __forceinline__ float act_tanh(float x) {
#if __has_builtin(__builtin_amdgcn_tanhf)
  return __builtin_amdgcn_tanhf(x);           // v_tanh_f32 (TRANS pipe)
#else
  return tanhf(x);
#endif
}

// xor-16 crosslane via v_permlanex16_b32 (VALU, no LDS): lanes 0-15 <-> 16-31, identity select.
static __device__ __forceinline__ unsigned xor16u(unsigned v) {
#if __has_builtin(__builtin_amdgcn_permlanex16)
  return __builtin_amdgcn_permlanex16(v, v, 0x76543210u, 0xFEDCBA98u, false, false);
#else
  return (unsigned)__shfl_xor((int)v, 16, 32);
#endif
}
static __device__ __forceinline__ float xor16f(float v) {
  return __builtin_bit_cast(float, xor16u(__builtin_bit_cast(unsigned, v)));
}

// D = A x B + C computed transposed: H^T(neurons x points) = W^T x Z^T + bias.
// Weights live as per-lane constants in WMMA A-layout; activations flow through
// B-layout (f16) / C-layout (f32). Elements whose matching A entry is zero (K>=10)
// are left as arbitrary *finite* values: 0 x finite == 0, so no zero-selects needed.
__global__ __launch_bounds__(256) void pairforce_mlp_kernel(
    const float* __restrict__ x,
    const float* __restrict__ W1, const float* __restrict__ b1,
    const float* __restrict__ W2, const float* __restrict__ b2,
    const float* __restrict__ W3, const float* __restrict__ b3,
    const float* __restrict__ W4, const float* __restrict__ b4,
    float* __restrict__ out, int npts)
{
  const int  lane = threadIdx.x & 31;
  const int  m    = lane & 15;
  const bool hi   = lane >= 16;

  // ---- per-lane constant weight tiles (set up once per wave) ----
  // Layer-1 A: f32 16x4 A-layout. lanes0-15: VGPR0=K0, VGPR1=K1 ; lanes16-31: VGPR0=K2, VGPR1=K3(pad0).
  v2f a1;
  a1.x = (m < 10) ? W1[(hi ? 2 : 0) * 10 + m] : 0.0f;
  a1.y = (m < 10 && !hi) ? W1[1 * 10 + m] : 0.0f;

  // Layer-2/3 A: f16 16x32 A-layout. lanes0-15 elems0..7=K0..7 ; lanes16-31 elems0..7=K8..15 ;
  // elems8..15=K16..31. All K>=10 entries are exactly zero (this is what makes B garbage safe).
  v16h a2 = {}, a3 = {};
#pragma unroll
  for (int j = 0; j < 8; ++j) {
    int k = hi ? (j + 8) : j;
    if (m < 10 && k < 10) {
      a2[j] = (_Float16)W2[k * 10 + m];   // W2^T[m,k]
      a3[j] = (_Float16)W3[k * 10 + m];
    }
  }

  // Bias as WMMA C operand: C-layout VGPR r -> M=r (lanes<16) / M=r+8 (lanes>=16).
  v8f cb1 = {}, cb2 = {}, cb3 = {};
  float w4l[8];
#pragma unroll
  for (int r = 0; r < 8; ++r) {
    int idx = r + (hi ? 8 : 0);
    bool v = idx < 10;
    cb1[r] = v ? b1[idx] : 0.0f;
    cb2[r] = v ? b2[idx] : 0.0f;
    cb3[r] = v ? b3[idx] : 0.0f;
    w4l[r] = v ? W4[idx] : 0.0f;
  }
  const float bias4 = b4[0];

  // C-layout f32 -> next-layer B-layout f16 with activation, packed-f16 math.
  // B 32x16 f16: lanes0-15 elems0..7 = K0..7, lanes16-31 elems0..7 = K8..15; elems8..15 = K16..31 (dead, A=0).
  // Own lane has neurons 0..7 (lanes<16) or 8..15 (lanes>=16); neurons 8,9 fetched via xor-16.
  auto pack_lrelu = [&](const v8f& h) -> v16h {
    v2h p0 = lrelu2h(pkrtz(h[0], h[1]));
    v2h p1 = lrelu2h(pkrtz(h[2], h[3]));
    v2h p2 = lrelu2h(pkrtz(h[4], h[5]));
    v2h p3 = lrelu2h(pkrtz(h[6], h[7]));
    unsigned o = xor16u(__builtin_bit_cast(unsigned, p0));   // neurons 8,9 (one packed shuffle)
    v8u gw = {__builtin_bit_cast(unsigned, p0), __builtin_bit_cast(unsigned, p1),
              __builtin_bit_cast(unsigned, p2), __builtin_bit_cast(unsigned, p3),
              o, 0u, 0u, 0u};
    return __builtin_bit_cast(v16h, gw);
  };
  auto pack_tanh = [&](const v8f& h) -> v16h {
    v2h p0 = pkrtz(act_tanh(h[0]), act_tanh(h[1]));
    v2h p1 = pkrtz(act_tanh(h[2]), act_tanh(h[3]));
    v2h p2 = pkrtz(act_tanh(h[4]), act_tanh(h[5]));
    v2h p3 = pkrtz(act_tanh(h[6]), act_tanh(h[7]));
    unsigned o = xor16u(__builtin_bit_cast(unsigned, p0));
    v8u gw = {__builtin_bit_cast(unsigned, p0), __builtin_bit_cast(unsigned, p1),
              __builtin_bit_cast(unsigned, p2), __builtin_bit_cast(unsigned, p3),
              o, 0u, 0u, 0u};
    return __builtin_bit_cast(v16h, gw);
  };

  const int nwaves = (int)(gridDim.x * (blockDim.x >> 5));
  const int wid    = (int)((blockIdx.x * blockDim.x + threadIdx.x) >> 5);

  for (int base = wid * 32; base < npts; base += nwaves * 32) {
    int p  = base + lane;                 // each lane owns one point; 2 WMMA tiles of 16
    int pc = p < npts ? p : npts - 1;
    const float* xp = x + pc * 6;
    v2f u0 = *(const v2f*)(xp + 0);       // x0,x1
    v2f u1 = *(const v2f*)(xp + 2);       // x2,x3
    v2f u2 = *(const v2f*)(xp + 4);       // x4,x5

    float dx  = u1.y - u0.x;              // x3-x0
    float dy  = u2.x - u0.y;              // x4-x1
    float dz  = u2.y - u1.x;              // x5-x2
    float sxy = dx * dx + dy * dy;
    float r2  = sxy + dz * dz;
    float zi  = u1.x;                     // x2
    float zj  = u2.y;                     // x5

    float o_zi  = xor16f(zi);
    float o_zj  = xor16f(zj);
    float o_sxy = xor16f(sxy);

    // Layer-1 B (f32 4x16): lanes0-15 = K0(z_i),K1(z_j) ; lanes16-31 = K2(sep_xy^2),K3(dead).
    v2f bt0, bt1;
    bt0.x = hi ? o_sxy : zi;   bt0.y = zj;     // tile0: points base+0..15
    bt1.x = hi ? sxy   : o_zi; bt1.y = o_zj;   // tile1: points base+16..31

    v8f h0 = __builtin_amdgcn_wmma_f32_16x16x4_f32(false, a1, false, bt0, (short)0, cb1, false, false);
    v8f h1 = __builtin_amdgcn_wmma_f32_16x16x4_f32(false, a1, false, bt1, (short)0, cb1, false, false);

    v16h g0 = pack_lrelu(h0);
    v16h g1 = pack_lrelu(h1);

    h0 = __builtin_amdgcn_wmma_f32_16x16x32_f16(false, a2, false, g0, (short)0, cb2, false, false);
    h1 = __builtin_amdgcn_wmma_f32_16x16x32_f16(false, a2, false, g1, (short)0, cb2, false, false);

    g0 = pack_tanh(h0);
    g1 = pack_tanh(h1);

    h0 = __builtin_amdgcn_wmma_f32_16x16x32_f16(false, a3, false, g0, (short)0, cb3, false, false);
    h1 = __builtin_amdgcn_wmma_f32_16x16x32_f16(false, a3, false, g1, (short)0, cb3, false, false);

    // Layer 4: F_amp = leaky(h3) . W4 + b4 ; lane halves reduced with one xor-16 each.
    float acc0 = 0.0f, acc1 = 0.0f;
#pragma unroll
    for (int r = 0; r < 8; ++r) {
      acc0 = fmaf(lrelu(h0[r]), w4l[r], acc0);
      acc1 = fmaf(lrelu(h1[r]), w4l[r], acc1);
    }
    float F0 = acc0 + xor16f(acc0) + bias4;
    float F1 = acc1 + xor16f(acc1) + bias4;
    float F  = hi ? F1 : F0;

    float inv = __builtin_amdgcn_rcpf(r2);
    v2f o; o.x = F * dx * inv; o.y = F * dy * inv;
    if (p < npts) *(v2f*)(out + 2 * p) = o;
  }
}

extern "C" void kernel_launch(void* const* d_in, const int* in_sizes, int n_in,
                              void* d_out, int out_size, void* d_ws, size_t ws_size,
                              hipStream_t stream) {
  const float* x  = (const float*)d_in[0];
  const float* W1 = (const float*)d_in[1];
  const float* b1 = (const float*)d_in[2];
  const float* W2 = (const float*)d_in[3];
  const float* b2 = (const float*)d_in[4];
  const float* W3 = (const float*)d_in[5];
  const float* b3 = (const float*)d_in[6];
  const float* W4 = (const float*)d_in[7];
  const float* b4 = (const float*)d_in[8];
  float* out = (float*)d_out;

  int npts = in_sizes[0] / 6;             // 512*16384 = 8,388,608
  dim3 block(256);                        // 8 wave32 per block
  dim3 grid(4096);                        // 32768 waves -> 8 iterations of 32 points each
  hipLaunchKernelGGL(pairforce_mlp_kernel, grid, block, 0, stream,
                     x, W1, b1, W2, b2, W3, b3, W4, b4, out, npts);
}
